// seq2seq_17386027615028
// MI455X (gfx1250) — matrix operand order repaired and checked by reading, not compiled, not run on hardware
//
#include <hip/hip_runtime.h>
#include <hip/hip_fp16.h>

// Sizes from the reference
#define L_SEQ 8192
#define HDIM  512
#define G3    1536   // 3*H
#define VDIM  128

typedef __attribute__((ext_vector_type(16))) _Float16 v16h;
typedef __attribute__((ext_vector_type(8)))  float    v8f;
typedef int ivec4 __attribute__((__vector_size__(16)));   // for async-LDS builtin

// Fast activations: v_rcp_f32 / v_exp_f32 / v_tanh_f32 instead of IEEE divides
// and ocml tanh. Gate math runs ~25M times; the precise div sequence seen in
// round-1 disasm (v_div_scale/v_div_fmas/v_div_fixup) is ~6x more VALU.
__device__ __forceinline__ float fast_rcp(float x) { return __builtin_amdgcn_rcpf(x); }
__device__ __forceinline__ float sigf(float x) { return fast_rcp(1.0f + __expf(-x)); }
__device__ __forceinline__ float tanh_fast(float x) {
#if __has_builtin(__builtin_amdgcn_tanhf)
    return __builtin_amdgcn_tanhf(x);
#else
    float t = __expf(2.0f * x);
    return 1.0f - 2.0f * fast_rcp(t + 1.0f);
#endif
}

#ifndef ASYNC_WEIGHT_LOAD
#define ASYNC_WEIGHT_LOAD 1
#endif
#if ASYNC_WEIGHT_LOAD && __has_builtin(__builtin_amdgcn_global_load_async_to_lds_b128)
#define HAVE_ASYNC_LDS 1
#else
#define HAVE_ASYNC_LDS 0
#endif

// ---------------------------------------------------------------------------
// WMMA GEMM:  out[M,N] = act( A' @ W^T + bias )
//   A : [M, K] f32 (MODE 1: rows remapped t->max(t-1,0), then relu)
//   W : [N, K] f32 row-major (B-column n == W row n)
//   MODE 0/1: store raw f32 (input projections, bias = bih)
//   MODE 2  : sigmoid epilogue (output projection -> d_out)
// Block: 256 threads = 8 waves; block tile M=32, N=128; K in 32-chunks
// (fully unrolled via template). Each wave computes two 16x16 M-subtiles of
// one 16-col strip, reusing its B operand -> 2 v_wmma per chunk per wave.
// Per-lane operand packing follows CDNA5 ISA 7.12.2 16-bit A/B layouts.
// ---------------------------------------------------------------------------
template <int KC, int MODE>
__global__ __launch_bounds__(256, 2)
void gru_wmma_gemm(const float* __restrict__ A, const float* __restrict__ W,
                   const float* __restrict__ bias, float* __restrict__ out,
                   int N, int lda)
{
    __shared__ _Float16 As[32 * 32];   // [m][k]
    __shared__ _Float16 Bs[128 * 32];  // [n][k]

    const int t    = threadIdx.x;
    const int wv   = t >> 5;        // wave id 0..7 -> 16-col sub-tile
    const int lane = t & 31;
    const int hh   = lane >> 4;     // lane half (0: lanes 0-15, 1: 16-31)
    const int mr   = lane & 15;
    const int m0   = blockIdx.x * 32;
    const int n0   = blockIdx.y * 128;

    v8f acc0 = {}, acc1 = {};

    for (int k0 = 0; k0 < KC; k0 += 32) {
        __syncthreads();
        // Stage A tile (32x32), f32 -> f16, optional shift+relu remap.
        for (int idx = t; idx < 32 * 32; idx += 256) {
            int r = idx >> 5, c = idx & 31;
            int row = m0 + r;
            float v;
            if (MODE == 1) {
                int sr = (row == 0) ? 0 : row - 1;
                v = A[(size_t)sr * lda + k0 + c];
                v = v > 0.0f ? v : 0.0f;
            } else {
                v = A[(size_t)row * lda + k0 + c];
            }
            As[idx] = (_Float16)v;
        }
        // Stage B tile (128x32) = rows n0..n0+127 of W.
        for (int idx = t; idx < 128 * 32; idx += 256) {
            int n = idx >> 5, c = idx & 31;
            Bs[idx] = (_Float16)W[(size_t)(n0 + n) * KC + k0 + c];
        }
        __syncthreads();

        v16h a0, a1, b;
#pragma unroll
        for (int j = 0; j < 16; ++j) {
            int ka = (j < 8 ? j : j + 8) + (hh ? 8 : 0);   // A: 16x32 MxK striping
            a0[j] = As[mr * 32 + ka];
            a1[j] = As[(16 + mr) * 32 + ka];
            int kb = (j < 8 ? j : j + 8) + (hh ? 8 : 0);   // B: 32x16 KxN striping
            b[j] = Bs[(wv * 16 + mr) * 32 + kb];
        }
        acc0 = __builtin_amdgcn_wmma_f32_16x16x32_f16(
            false, a0, false, b, (short)0, acc0, false, false);
        acc1 = __builtin_amdgcn_wmma_f32_16x16x32_f16(
            false, a1, false, b, (short)0, acc1, false, false);
    }

    // Epilogue: element e of v8f is (M = m0+e+8*hh, N = lane&15 of strip).
    const int col = n0 + wv * 16 + mr;
    const float bv = bias[col];
#pragma unroll
    for (int e = 0; e < 8; ++e) {
        int m = m0 + e + hh * 8;
        float x0 = acc0[e] + bv;
        float x1 = acc1[e] + bv;
        if (MODE == 2) { x0 = sigf(x0); x1 = sigf(x1); }
        out[(size_t)m * N + col]        = x0;
        out[(size_t)(m + 16) * N + col] = x1;
    }
}

// ---------------------------------------------------------------------------
// Persistent grid-cooperative 2-layer GRU scan (one grid barrier / step).
//   Grid = 48 WGs x 256 threads. WGs 0..15: layer 0 (32 units each).
//   WGs 16..47: layer 1 (16 units each), pipelined one step behind layer 0.
//   Recurrent weights live in LDS (~195 KB/WG of the 320 KB WGP LDS), loaded
//   once (via gfx1250 async-DMA global->LDS when available) and reused for
//   all 8192 steps. h-states double-buffered in L2 via device-scope atomics.
//   LDS weight rows use a 520-float stride; lanes read i = seg + nseg*ii,
//   which puts all 32 lanes of a wave on 32 distinct banks.
// ---------------------------------------------------------------------------
#define WROW_STRIDE 520

__global__ __launch_bounds__(256, 1)
void gru_seq(const float* __restrict__ gx,    // [L, 1536] x@Wih0^T + bih0
             const float* __restrict__ Whh0, const float* __restrict__ bhh0,
             const float* __restrict__ Wih1, const float* __restrict__ bih1,
             const float* __restrict__ Whh1, const float* __restrict__ bhh1,
             float* __restrict__ h0buf,       // 2 x 512 (double buffered)
             float* __restrict__ h1buf,       // 2 x 512
             float* __restrict__ ys,          // [L, 512] top outputs
             int* __restrict__ bar,
             int Lsteps, int storeYs)
{
    extern __shared__ float sm[];
    float* wS = sm;                       // 96 rows * 520
    float* hA = sm + 96 * WROW_STRIDE;    // 512 (h0 / c0)
    float* hB = hA + 512;                 // 512 (h1, layer-1 WGs only)

    const int t   = threadIdx.x;
    const int g   = blockIdx.x;
    const int NWG = gridDim.x;
    const bool isL1 = (g >= 16);

    // ---- One-time weight load into LDS (amortized over 8192 steps) ----
#if HAVE_ASYNC_LDS
    {
        // 96 rows x 512 floats, copied as per-lane b128 async DMA transfers
        // (ASYNCcnt-tracked; bulk contiguous, no coherence hazard).
        const int u0 = isL1 ? (g - 16) * 16 : g * 32;
        for (int idx = t; idx < 96 * 128; idx += 256) {
            int rl = idx >> 7, c4 = idx & 127;   // row-local, 16B chunk
            const float* Wsrc;
            int grow;
            if (!isL1) {
                int ui = rl / 3, gate = rl % 3;
                Wsrc = Whh0;
                grow = gate * 512 + u0 + ui;
            } else {
                int ui = rl / 6, p = rl % 6;
                Wsrc = (p < 3) ? Wih1 : Whh1;
                int gate = (p < 3) ? p : p - 3;
                grow = gate * 512 + u0 + ui;
            }
            const float* gp = Wsrc + (size_t)grow * 512 + c4 * 4;
            float* lp = wS + rl * WROW_STRIDE + c4 * 4;
            __builtin_amdgcn_global_load_async_to_lds_b128(
                (__attribute__((address_space(1))) ivec4*)gp,
                (__attribute__((address_space(3))) ivec4*)lp, 0, 0);
        }
#if __has_builtin(__builtin_amdgcn_s_wait_asynccnt)
        __builtin_amdgcn_s_wait_asynccnt(0);
#else
        asm volatile("s_wait_asynccnt 0" ::: "memory");
#endif
    }
#else
    if (!isL1) {
        const int u0 = g * 32;
        for (int idx = t; idx < 96 * 512; idx += 256) {
            int rl = idx >> 9, i = idx & 511;     // rl = 3*ui + gate
            int ui = rl / 3, gate = rl % 3;
            wS[rl * WROW_STRIDE + i] = Whh0[(size_t)(gate * 512 + u0 + ui) * 512 + i];
        }
    } else {
        const int u0 = (g - 16) * 16;
        for (int idx = t; idx < 96 * 512; idx += 256) {
            int rl = idx >> 9, i = idx & 511;     // rl = 6*ui + p
            int ui = rl / 6, p = rl % 6;
            const float* Wsrc = (p < 3) ? Wih1 : Whh1;
            int gate = (p < 3) ? p : p - 3;
            wS[rl * WROW_STRIDE + i] = Wsrc[(size_t)(gate * 512 + u0 + ui) * 512 + i];
        }
    }
#endif
    __syncthreads();

    for (int k = 0; k <= Lsteps; ++k) {
        if (!isL1) {
            if (k < Lsteps) {
                const float* hp = h0buf + (k & 1) * 512;   // state before step k
                for (int i = t; i < 512; i += 256)
                    hA[i] = __hip_atomic_load(&hp[i], __ATOMIC_RELAXED,
                                              __HIP_MEMORY_SCOPE_AGENT);
                __syncthreads();

                const int ui = t >> 3, seg = t & 7;
                float ar = 0.f, az = 0.f, an = 0.f;
                const float* wr = wS + (3 * ui + 0) * WROW_STRIDE;
                const float* wz = wS + (3 * ui + 1) * WROW_STRIDE;
                const float* wn = wS + (3 * ui + 2) * WROW_STRIDE;
#pragma unroll 8
                for (int ii = 0; ii < 64; ++ii) {
                    int i = seg + 8 * ii;          // bank-conflict-free stride
                    float hv = hA[i];
                    ar = __builtin_fmaf(wr[i], hv, ar);
                    az = __builtin_fmaf(wz[i], hv, az);
                    an = __builtin_fmaf(wn[i], hv, an);
                }
#pragma unroll
                for (int o = 4; o > 0; o >>= 1) {  // reduce 8 consecutive lanes
                    ar += __shfl_down(ar, o, 32);
                    az += __shfl_down(az, o, 32);
                    an += __shfl_down(an, o, 32);
                }
                if (seg == 0) {
                    const int j = g * 32 + ui;
                    const float* gxt = gx + (size_t)k * G3;
                    float r = sigf(gxt[j]        + ar + bhh0[j]);
                    float z = sigf(gxt[512 + j]  + az + bhh0[512 + j]);
                    float n = tanh_fast(gxt[1024 + j] + r * (an + bhh0[1024 + j]));
                    float hn = (1.f - z) * n + z * hA[j];
                    __hip_atomic_store(&h0buf[((k + 1) & 1) * 512 + j], hn,
                                       __ATOMIC_RELAXED, __HIP_MEMORY_SCOPE_AGENT);
                }
            }
        } else {
            if (k >= 1) {
                const int t1 = k - 1;              // layer 1 processes step k-1
                const float* c0p = h0buf + (k & 1) * 512;        // h0 after k-1
                const float* h1p = h1buf + ((k - 1) & 1) * 512;  // h1 before k-1
                for (int i = t; i < 512; i += 256) {
                    hA[i] = __hip_atomic_load(&c0p[i], __ATOMIC_RELAXED,
                                              __HIP_MEMORY_SCOPE_AGENT);
                    hB[i] = __hip_atomic_load(&h1p[i], __ATOMIC_RELAXED,
                                              __HIP_MEMORY_SCOPE_AGENT);
                }
                __syncthreads();

                const int ui = t >> 4, seg = t & 15;
                float xr = 0.f, xz = 0.f, xn = 0.f, hr = 0.f, hz = 0.f, hn = 0.f;
                const float* w = wS + 6 * ui * WROW_STRIDE;
#pragma unroll 8
                for (int ii = 0; ii < 32; ++ii) {
                    int i = seg + 16 * ii;
                    float cv = hA[i], hv = hB[i];
                    xr = __builtin_fmaf(w[0 * WROW_STRIDE + i], cv, xr);
                    xz = __builtin_fmaf(w[1 * WROW_STRIDE + i], cv, xz);
                    xn = __builtin_fmaf(w[2 * WROW_STRIDE + i], cv, xn);
                    hr = __builtin_fmaf(w[3 * WROW_STRIDE + i], hv, hr);
                    hz = __builtin_fmaf(w[4 * WROW_STRIDE + i], hv, hz);
                    hn = __builtin_fmaf(w[5 * WROW_STRIDE + i], hv, hn);
                }
#pragma unroll
                for (int o = 8; o > 0; o >>= 1) {  // reduce 16 consecutive lanes
                    xr += __shfl_down(xr, o, 32);
                    xz += __shfl_down(xz, o, 32);
                    xn += __shfl_down(xn, o, 32);
                    hr += __shfl_down(hr, o, 32);
                    hz += __shfl_down(hz, o, 32);
                    hn += __shfl_down(hn, o, 32);
                }
                if (seg == 0) {
                    const int j = (g - 16) * 16 + ui;
                    float r = sigf(xr + bih1[j]        + hr + bhh1[j]);
                    float z = sigf(xz + bih1[512 + j]  + hz + bhh1[512 + j]);
                    float n = tanh_fast(xn + bih1[1024 + j] + r * (hn + bhh1[1024 + j]));
                    float hv = (1.f - z) * n + z * hB[j];
                    __hip_atomic_store(&h1buf[(k & 1) * 512 + j], hv,
                                       __ATOMIC_RELAXED, __HIP_MEMORY_SCOPE_AGENT);
                    if (storeYs) ys[(size_t)t1 * 512 + j] = hv;
                }
            }
        }

        // ---- Grid barrier: monotonic counter, one per iteration ----
        __syncthreads();
        if (t == 0) {
            __threadfence();
            __hip_atomic_fetch_add(bar, 1, __ATOMIC_ACQ_REL, __HIP_MEMORY_SCOPE_AGENT);
            const int tgt = NWG * (k + 1);
            while (__hip_atomic_load(bar, __ATOMIC_ACQUIRE,
                                     __HIP_MEMORY_SCOPE_AGENT) < tgt)
                __builtin_amdgcn_s_sleep(1);
        }
        __syncthreads();
    }
}

__global__ void gru_init(float* h0, float* h1, int* bar, int zeroH)
{
    int t = blockIdx.x * blockDim.x + threadIdx.x;
    if (t < 4) bar[t] = 0;
    if (zeroH && t < 1024) { h0[t] = 0.0f; h1[t] = 0.0f; }
}

// ---------------------------------------------------------------------------
extern "C" void kernel_launch(void* const* d_in, const int* in_sizes, int n_in,
                              void* d_out, int out_size, void* d_ws, size_t ws_size,
                              hipStream_t stream)
{
    (void)in_sizes; (void)n_in; (void)out_size; (void)ws_size;

    const float* src   = (const float*)d_in[0];
    const float* trg   = (const float*)d_in[1];
    const float* eWih0 = (const float*)d_in[2];
    const float* eWhh0 = (const float*)d_in[3];
    const float* ebih0 = (const float*)d_in[4];
    const float* ebhh0 = (const float*)d_in[5];
    const float* eWih1 = (const float*)d_in[6];
    const float* eWhh1 = (const float*)d_in[7];
    const float* ebih1 = (const float*)d_in[8];
    const float* ebhh1 = (const float*)d_in[9];
    const float* dWih0 = (const float*)d_in[10];
    const float* dWhh0 = (const float*)d_in[11];
    const float* dbih0 = (const float*)d_in[12];
    const float* dbhh0 = (const float*)d_in[13];
    const float* dWih1 = (const float*)d_in[14];
    const float* dWhh1 = (const float*)d_in[15];
    const float* dbih1 = (const float*)d_in[16];
    const float* dbhh1 = (const float*)d_in[17];
    const float* Wout  = (const float*)d_in[18];
    const float* bout  = (const float*)d_in[19];
    float* out = (float*)d_out;

    // Workspace (f32): gx_enc[L*3H] gx_dec[L*3H] ys[L*H] h0[1024] h1[1024] bar
    float* gx_enc = (float*)d_ws;
    float* gx_dec = gx_enc + (size_t)L_SEQ * G3;
    float* ys     = gx_dec + (size_t)L_SEQ * G3;
    float* h0b    = ys + (size_t)L_SEQ * HDIM;
    float* h1b    = h0b + 1024;
    int*   bar    = (int*)(h1b + 1024);

    const size_t seqLds = (96 * WROW_STRIDE + 1024) * sizeof(float); // ~199 KB

    dim3 blk(256);

    // 1) Encoder layer-0 input projection: gx_enc = src @ eWih0^T + ebih0
    gru_wmma_gemm<VDIM, 0><<<dim3(L_SEQ / 32, G3 / 128), blk, 0, stream>>>(
        src, eWih0, ebih0, gx_enc, G3, VDIM);

    // 2) Decoder layer-0 input projection on relu(shifted trg)
    gru_wmma_gemm<VDIM, 1><<<dim3(L_SEQ / 32, G3 / 128), blk, 0, stream>>>(
        trg, dWih0, dbih0, gx_dec, G3, VDIM);

    // 3) Encoder scan (h init = 0)
    gru_init<<<1, 1024, 0, stream>>>(h0b, h1b, bar, 1);
    gru_seq<<<48, 256, seqLds, stream>>>(
        gx_enc, eWhh0, ebhh0, eWih1, ebih1, eWhh1, ebhh1,
        h0b, h1b, ys /*unused*/, bar, L_SEQ, 0);

    // 4) Decoder scan (h init = encoder finals, already in buffer 0)
    gru_init<<<1, 1024, 0, stream>>>(h0b, h1b, bar, 0);
    gru_seq<<<48, 256, seqLds, stream>>>(
        gx_dec, dWhh0, dbhh0, dWih1, dbih1, dWhh1, dbhh1,
        h0b, h1b, ys, bar, L_SEQ, 1);

    // 5) Output projection + sigmoid: out = sigmoid(ys @ Wout^T + bout)
    gru_wmma_gemm<HDIM, 2><<<dim3(L_SEQ / 32, VDIM / 128), blk, 0, stream>>>(
        ys, Wout, bout, out, VDIM, HDIM);
}
